// MultiHeadAttention_23510650979027
// MI455X (gfx1250) — compile-verified, matrix operand order checked by reading, not run on hardware
//
#include <hip/hip_runtime.h>

#define BATCH 64
#define SEQL  256
#define DIMS  1024
#define NHEAD 16
#define HDIM  64

typedef __attribute__((ext_vector_type(16))) __bf16       v16bf;
typedef __attribute__((ext_vector_type(8)))  float        v8f;
typedef __attribute__((ext_vector_type(4)))  float        f32x4;
typedef __attribute__((ext_vector_type(4)))  unsigned int u32x4;
typedef __attribute__((ext_vector_type(2)))  unsigned int u32x2;

union U16B {
    u32x4 q[2];
    v16bf v;
};

static __device__ __forceinline__ v8f wmma_bf16(v16bf a, v16bf b, v8f c) {
    // (neg_a, A, neg_b, B, c_mod, C, reuse_a, reuse_b)
    return __builtin_amdgcn_wmma_f32_16x16x32_bf16(false, a, false, b, (short)0, c,
                                                   false, false);
}

// ---------------------------------------------------------------------------
// fp32 -> bf16 conversion (vectorized: float4 in, packed 4x bf16 out)
// ---------------------------------------------------------------------------
__global__ void cvt_f32_bf16(const float* __restrict__ in, __bf16* __restrict__ out,
                             int n4) {
    int i = blockIdx.x * blockDim.x + threadIdx.x;
    if (i < n4) {
        f32x4 f = ((const f32x4*)in)[i];
        union { unsigned short s[4]; u32x2 q; } pk;
#pragma unroll
        for (int j = 0; j < 4; ++j) {
            __bf16 b = (__bf16)f[j];
            pk.s[j]  = __builtin_bit_cast(unsigned short, b);
        }
        ((u32x2*)out)[i] = pk.q;
    }
}

// ---------------------------------------------------------------------------
// QKV projection (y = x @ W^T) + RoPE on Q/K.
// Block = 8 waves sharing one (head, proj): the 64x1024 weight tile is staged
// through LDS in double-buffered 64x64 slabs (8 KB each) -> 8x less global
// weight traffic; B-operands come from ds_load_b128.
// One wave = 16 rows x 64 cols. V written transposed (B,H,Hd,S).
// ---------------------------------------------------------------------------
__global__ __launch_bounds__(256) void gemm_qkv_rope(
    const __bf16* __restrict__ xb, const __bf16* __restrict__ wq,
    const __bf16* __restrict__ wk, const __bf16* __restrict__ wv,
    __bf16* __restrict__ qo, __bf16* __restrict__ ko, __bf16* __restrict__ vto) {
    __shared__ u32x4 Wlds[2][512];  // 2 x (64 rows x 8 u32x4) = 16 KB

    const int tid  = threadIdx.x;
    const int wave = tid >> 5;
    const int lane = tid & 31;
    const int ln   = lane & 15;
    const int hf   = lane >> 4;
    const int rowTile = blockIdx.x * 8 + wave;  // 0..1023 (B*S/16)
    const int h    = blockIdx.y;
    const int proj = blockIdx.z;
    const int r0   = rowTile * 16;
    const int rowBase = h * HDIM;

    const __bf16* W = (proj == 0) ? wq : (proj == 1) ? wk : wv;

    // cooperative W slab loader: thread -> 2 u32x4 (32B)
    const int lidx = tid * 2;
    const int lrow = lidx >> 3;
    const int lc4  = lidx & 7;
    const __bf16* wlp = W + (size_t)(rowBase + lrow) * DIMS + lc4 * 8;

    v8f acc[4] = {};
    const __bf16* xrow = xb + (size_t)(r0 + ln) * DIMS + hf * 8;

    // preload slab 0
    {
        const u32x4* s = (const u32x4*)wlp;
        Wlds[0][lidx]     = s[0];
        Wlds[0][lidx + 1] = s[1];
    }
    __syncthreads();

    for (int step = 0; step < 16; ++step) {
        const int cur = step & 1;
        const int kk  = step * 64;
        if (step + 1 < 16) {  // fill other buffer while computing
            const u32x4* s = (const u32x4*)(wlp + kk + 64);
            Wlds[cur ^ 1][lidx]     = s[0];
            Wlds[cur ^ 1][lidx + 1] = s[1];
        }
        if (step + 2 < 16)  // CDNA5 global_prefetch_b8 for the +2 slab
            __builtin_prefetch((const void*)(wlp + kk + 128), 0, 1);

        U16B A0, A1;
        const u32x4* xp = (const u32x4*)(xrow + kk);
        A0.q[0] = xp[0];
        A0.q[1] = xp[2];
        A1.q[0] = xp[4];
        A1.q[1] = xp[6];
#pragma unroll
        for (int t = 0; t < 4; ++t) {
            const int rb = (16 * t + ln) * 8;
            U16B B0, B1;
            B0.q[0] = Wlds[cur][rb + hf];
            B0.q[1] = Wlds[cur][rb + hf + 2];
            B1.q[0] = Wlds[cur][rb + 4 + hf];
            B1.q[1] = Wlds[cur][rb + 4 + hf + 2];
            acc[t] = wmma_bf16(A0.v, B0.v, acc[t]);
            acc[t] = wmma_bf16(A1.v, B1.v, acc[t]);
        }
        __syncthreads();
    }

    const int b  = r0 / SEQL;
    const int sb = r0 % SEQL;

    if (proj < 2) {
        // RoPE: col d = 16t+ln pairs with d+32 (tile t+2), same lane.
#pragma unroll
        for (int t = 0; t < 2; ++t) {
            const int   i    = t * 16 + ln;  // freq index 0..31
            const float invf = __powf(10000.f, -(float)i * (1.f / 32.f));
#pragma unroll
            for (int r = 0; r < 8; ++r) {
                const int pos = sb + r + hf * 8;
                float     sn, cs;
                __sincosf((float)pos * invf, &sn, &cs);
                float a0 = acc[t][r], a2 = acc[t + 2][r];
                acc[t][r]     = a0 * cs - a2 * sn;
                acc[t + 2][r] = a2 * cs + a0 * sn;
            }
        }
        __bf16* dst = (proj == 0 ? qo : ko) + ((size_t)b * NHEAD + h) * SEQL * HDIM;
#pragma unroll
        for (int t = 0; t < 4; ++t)
#pragma unroll
            for (int r = 0; r < 8; ++r)
                dst[(size_t)(sb + r + hf * 8) * HDIM + 16 * t + ln] = (__bf16)acc[t][r];
    } else {
        // V transposed: Vt[b,h,hd,s]; rows r are consecutive s -> 16B store
        __bf16* dst = vto + ((size_t)b * NHEAD + h) * HDIM * SEQL;
#pragma unroll
        for (int t = 0; t < 4; ++t) {
            union { unsigned short s[8]; u32x4 qv; } pk;
#pragma unroll
            for (int r = 0; r < 8; ++r) {
                __bf16 bv = (__bf16)acc[t][r];
                pk.s[r]   = __builtin_bit_cast(unsigned short, bv);
            }
            *(u32x4*)(dst + (size_t)(16 * t + ln) * SEQL + sb + hf * 8) = pk.qv;
        }
    }
}

// ---------------------------------------------------------------------------
// Attention: one block per (b,h), 16 waves x 16 queries. Computes S^T = K x Q^T
// (softmax is per-lane + one shfl_xor 16), then O^T = V^T x P^T where the
// normalized S^T accumulators pack directly into the B-operand (no transpose).
// ---------------------------------------------------------------------------
__global__ __launch_bounds__(512) void attn_kernel(
    const __bf16* __restrict__ q, const __bf16* __restrict__ k,
    const __bf16* __restrict__ vt, const int* __restrict__ mask,
    __bf16* __restrict__ x2) {
    __shared__ u32x4 Klds[2048];  // 256 keys x 64 bf16  (32 KB)
    __shared__ u32x4 Vlds[2048];  // 64 hd  x 256 bf16   (32 KB)

    const int bh = blockIdx.x;
    const int b  = bh >> 4;
    const int h  = bh & 15;
    {
        const u32x4* kg = (const u32x4*)(k + (size_t)bh * SEQL * HDIM);
        const u32x4* vg = (const u32x4*)(vt + (size_t)bh * HDIM * SEQL);
        for (int i = threadIdx.x; i < 2048; i += 512) {
            Klds[i] = kg[i];
            Vlds[i] = vg[i];
        }
    }
    __syncthreads();

    const int wave = threadIdx.x >> 5;
    const int lane = threadIdx.x & 31;
    const int ln   = lane & 15;
    const int hf   = lane >> 4;
    const int q0   = wave * 16;

    // Q tile as B-operand (= A-layout of Q rows), 2 K-chunks of 32
    U16B Qc[2];
#pragma unroll
    for (int c = 0; c < 2; ++c) {
        const u32x4* qp =
            (const u32x4*)(q + ((size_t)bh * SEQL + q0 + ln) * HDIM + 32 * c + hf * 8);
        Qc[c].q[0] = qp[0];
        Qc[c].q[1] = qp[2];
    }

    // S^T tiles: rows = keys, cols = queries (lane ln holds query q0+ln)
    v8f st[16];
#pragma unroll 4
    for (int n = 0; n < 16; ++n) {
        v8f       a   = {};
        const int key = 16 * n + ln;
#pragma unroll
        for (int c = 0; c < 2; ++c) {
            U16B      Kc;
            const int bi = key * 8 + 4 * c + hf;  // 8 u32x4 per key row
            Kc.q[0]      = Klds[bi];
            Kc.q[1]      = Klds[bi + 2];
            a            = wmma_bf16(Kc.v, Qc[c].v, a);
        }
        st[n] = a;
    }

    // softmax over keys: per-lane partial (half the keys) + xor-16 combine
    const float scale = 0.125f;  // Hd^-0.5
    float       mx    = -3.0e38f;
#pragma unroll
    for (int n = 0; n < 16; ++n)
#pragma unroll
        for (int r = 0; r < 8; ++r) {
            const int key = 16 * n + r + hf * 8;
            float     s   = st[n][r] * scale;
            if (mask[b * SEQL + key] == 0) s = -3.0e38f;
            st[n][r] = s;
            mx       = fmaxf(mx, s);
        }
    mx        = fmaxf(mx, __shfl_xor(mx, 16, 32));
    float sum = 0.f;
#pragma unroll
    for (int n = 0; n < 16; ++n)
#pragma unroll
        for (int r = 0; r < 8; ++r) {
            float p  = __expf(st[n][r] - mx);
            st[n][r] = p;
            sum += p;
        }
    sum             = sum + __shfl_xor(sum, 16, 32);
    const float rcp = 1.f / sum;
#pragma unroll
    for (int n = 0; n < 16; ++n) st[n] = st[n] * rcp;

    // O^T = V^T x P^T ; P^T accumulators pack straight into the B-operand
    v8f o[4] = {};
#pragma unroll
    for (int c = 0; c < 8; ++c) {  // 32-key chunks
        U16B Pb;
#pragma unroll
        for (int e = 0; e < 8; ++e) {
            Pb.v[e]     = (__bf16)st[2 * c][e];
            Pb.v[8 + e] = (__bf16)st[2 * c + 1][e];
        }
#pragma unroll
        for (int tm = 0; tm < 4; ++tm) {
            const int hd = 16 * tm + ln;
            const int bi = hd * 32 + 4 * c + hf;  // 32 u32x4 per hd row
            U16B      Va;
            Va.q[0] = Vlds[bi];
            Va.q[1] = Vlds[bi + 2];
            o[tm]   = wmma_bf16(Va.v, Pb.v, o[tm]);
        }
    }

    // store O (bf16, (B,S,D) layout): lane = query, VGPRs = 8 consecutive hd
#pragma unroll
    for (int tm = 0; tm < 4; ++tm) {
        union { unsigned short s[8]; u32x4 qv; } pk;
#pragma unroll
        for (int r = 0; r < 8; ++r) {
            __bf16 bv = (__bf16)o[tm][r];
            pk.s[r]   = __builtin_bit_cast(unsigned short, bv);
        }
        *(u32x4*)(x2 + ((size_t)b * SEQL + q0 + ln) * DIMS + h * HDIM + 16 * tm +
                  hf * 8) = pk.qv;
    }
}

// ---------------------------------------------------------------------------
// Output projection: out = x2 @ Wo^T, fp32 stores. Same LDS-staged weights.
// ---------------------------------------------------------------------------
__global__ __launch_bounds__(256) void gemm_out(const __bf16* __restrict__ x2,
                                                const __bf16* __restrict__ wo,
                                                float* __restrict__ out) {
    __shared__ u32x4 Wlds[2][512];

    const int tid     = threadIdx.x;
    const int wave    = tid >> 5;
    const int lane    = tid & 31;
    const int ln      = lane & 15;
    const int hf      = lane >> 4;
    const int rowTile = blockIdx.x * 8 + wave;
    const int cg      = blockIdx.y;  // 64-column group
    const int r0      = rowTile * 16;
    const int rowBase = cg * 64;

    const int lidx = tid * 2;
    const int lrow = lidx >> 3;
    const int lc4  = lidx & 7;
    const __bf16* wlp = wo + (size_t)(rowBase + lrow) * DIMS + lc4 * 8;

    v8f           acc[4] = {};
    const __bf16* xrow   = x2 + (size_t)(r0 + ln) * DIMS + hf * 8;

    {
        const u32x4* s = (const u32x4*)wlp;
        Wlds[0][lidx]     = s[0];
        Wlds[0][lidx + 1] = s[1];
    }
    __syncthreads();

    for (int step = 0; step < 16; ++step) {
        const int cur = step & 1;
        const int kk  = step * 64;
        if (step + 1 < 16) {
            const u32x4* s = (const u32x4*)(wlp + kk + 64);
            Wlds[cur ^ 1][lidx]     = s[0];
            Wlds[cur ^ 1][lidx + 1] = s[1];
        }
        if (step + 2 < 16)
            __builtin_prefetch((const void*)(wlp + kk + 128), 0, 1);

        U16B A0, A1;
        const u32x4* xp = (const u32x4*)(xrow + kk);
        A0.q[0] = xp[0];
        A0.q[1] = xp[2];
        A1.q[0] = xp[4];
        A1.q[1] = xp[6];
#pragma unroll
        for (int t = 0; t < 4; ++t) {
            const int rb = (16 * t + ln) * 8;
            U16B B0, B1;
            B0.q[0] = Wlds[cur][rb + hf];
            B0.q[1] = Wlds[cur][rb + hf + 2];
            B1.q[0] = Wlds[cur][rb + 4 + hf];
            B1.q[1] = Wlds[cur][rb + 4 + hf + 2];
            acc[t] = wmma_bf16(A0.v, B0.v, acc[t]);
            acc[t] = wmma_bf16(A1.v, B1.v, acc[t]);
        }
        __syncthreads();
    }
#pragma unroll
    for (int t = 0; t < 4; ++t)
#pragma unroll
        for (int r = 0; r < 8; ++r)
            out[(size_t)(r0 + r + hf * 8) * DIMS + cg * 64 + 16 * t + ln] = acc[t][r];
}

// ---------------------------------------------------------------------------
extern "C" void kernel_launch(void* const* d_in, const int* in_sizes, int n_in,
                              void* d_out, int out_size, void* d_ws, size_t ws_size,
                              hipStream_t stream) {
    (void)in_sizes; (void)n_in; (void)out_size; (void)ws_size;
    const float* x    = (const float*)d_in[0];
    const int*   mask = (const int*)d_in[1];
    const float* Wq   = (const float*)d_in[2];
    const float* Wk   = (const float*)d_in[3];
    const float* Wv   = (const float*)d_in[4];
    const float* Wo   = (const float*)d_in[5];

    const size_t NX = (size_t)BATCH * SEQL * DIMS;  // 16M elements
    const size_t NW = (size_t)DIMS * DIMS;          // 1M elements

    char*   ws  = (char*)d_ws;
    size_t  off = 0;
    __bf16* xb  = (__bf16*)(ws + off); off += NX * 2;
    __bf16* wqb = (__bf16*)(ws + off); off += NW * 2;
    __bf16* wkb = (__bf16*)(ws + off); off += NW * 2;
    __bf16* wvb = (__bf16*)(ws + off); off += NW * 2;
    __bf16* wob = (__bf16*)(ws + off); off += NW * 2;
    __bf16* qb  = (__bf16*)(ws + off); off += NX * 2;
    __bf16* kb  = (__bf16*)(ws + off); off += NX * 2;
    __bf16* vtb = (__bf16*)(ws + off); off += NX * 2;
    __bf16* x2b = (__bf16*)(ws + off); off += NX * 2;

    cvt_f32_bf16<<<(int)((NX / 4 + 255) / 256), 256, 0, stream>>>(x, xb, (int)(NX / 4));
    cvt_f32_bf16<<<(int)((NW / 4 + 255) / 256), 256, 0, stream>>>(Wq, wqb, (int)(NW / 4));
    cvt_f32_bf16<<<(int)((NW / 4 + 255) / 256), 256, 0, stream>>>(Wk, wkb, (int)(NW / 4));
    cvt_f32_bf16<<<(int)((NW / 4 + 255) / 256), 256, 0, stream>>>(Wv, wvb, (int)(NW / 4));
    cvt_f32_bf16<<<(int)((NW / 4 + 255) / 256), 256, 0, stream>>>(Wo, wob, (int)(NW / 4));

    // (B*S/16)/8 row-tile blocks x heads x {Q,K,V}
    gemm_qkv_rope<<<dim3(128, NHEAD, 3), 256, 0, stream>>>(xb, wqb, wkb, wvb, qb, kb,
                                                           vtb);
    // one block per (b,h)
    attn_kernel<<<dim3(BATCH * NHEAD), 512, 0, stream>>>(qb, kb, vtb, mask, x2b);
    // final projection, fp32 out
    gemm_out<<<dim3(128, DIMS / 64), 256, 0, stream>>>(x2b, wob, (float*)d_out);
}
